// GCN_7052336300581
// MI455X (gfx1250) — matrix operand order repaired and checked by reading, not compiled, not run on hardware
//
#include <hip/hip_runtime.h>
#include <math.h>

// GCN(512x512 GoL) + MLP(1,200,1,8 layers) for MI455X gfx1250 (wave32, WMMA).
#define HID   200
#define LDH   224   // padded K stride for activations: 7 * 32
#define LDW   208   // padded output-feature count: 13 * 16
#define KT    7     // k tiles of 32
#define NT    13    // n tiles of 16
#define SPLIT 112   // weight columns staged in LDS per stage (LDS = 224*112*2 = 50176 B)

typedef __attribute__((ext_vector_type(16))) _Float16 v16h;
typedef __attribute__((ext_vector_type(8)))  _Float16 v8h;
typedef __attribute__((ext_vector_type(8)))  float    v8f;
typedef int v4i_vs __attribute__((vector_size(4 * sizeof(int))));

// ---- CDNA5 async global->LDS copy (guarded; falls back to load+store) ----

static __device__ __forceinline__ void copy16_to_lds(const _Float16* g, _Float16* l) {
#if __has_builtin(__builtin_amdgcn_global_load_async_to_lds_b128)
  __builtin_amdgcn_global_load_async_to_lds_b128(
      (v4i_vs*)const_cast<_Float16*>(g), (v4i_vs*)l, 0, 0);
#else
  *(v8h*)l = *(const v8h*)g;
#endif
}

static __device__ __forceinline__ void wait_async0() {
#if __has_builtin(__builtin_amdgcn_s_wait_asynccnt)
  __builtin_amdgcn_s_wait_asynccnt(0);
#else
  asm volatile("s_wait_asynccnt 0x0" ::: "memory");
#endif
}

// ---------------- aggregation (bandwidth-bound, atomics) ----------------

__global__ void k_deg_init(float* __restrict__ deg, int n) {
  int i = blockIdx.x * blockDim.x + threadIdx.x;
  if (i < n) deg[i] = 1.0f;                 // self-loop contributes 1
}

__global__ void k_count(const long long* __restrict__ ei, float* __restrict__ deg, int E) {
  int e = blockIdx.x * blockDim.x + threadIdx.x;
  if (e < E) {
    int d = (int)ei[(long long)E + e];      // row 1 = dst
    atomicAdd(&deg[d], 1.0f);
  }
}

__global__ void k_norm(const float* __restrict__ x, const float* __restrict__ deg,
                       float* __restrict__ inv, float* __restrict__ agg, int n) {
  int i = blockIdx.x * blockDim.x + threadIdx.x;
  if (i < n) {
    float dv = deg[i];
    dv = dv < 1.0f ? 1.0f : dv;
    float iv = rsqrtf(dv);
    inv[i] = iv;
    agg[i] = x[i] * iv * iv;                // self-loop message
  }
}

__global__ void k_edge(const long long* __restrict__ ei, const float* __restrict__ x,
                       const float* __restrict__ inv, float* __restrict__ agg, int E) {
  int e = blockIdx.x * blockDim.x + threadIdx.x;
  if (e < E) {
    int s = (int)ei[e];
    int d = (int)ei[(long long)E + e];
    atomicAdd(&agg[d], x[s] * inv[s] * inv[d]);
  }
}

// ---------------- weight prep: f32 [200,200] -> f16 padded [224,208] ----------------

__global__ void k_prep_w(const float* __restrict__ W, _Float16* __restrict__ Wp) {
  int idx = blockIdx.x * blockDim.x + threadIdx.x;
  if (idx < LDH * LDW) {
    int r = idx / LDW, c = idx % LDW;
    float v = (r < HID && c < HID) ? W[r * HID + c] : 0.0f;
    Wp[idx] = (_Float16)v;
  }
}

__global__ void k_prep_b(const float* __restrict__ b, float* __restrict__ Bp) {
  int i = threadIdx.x;
  if (i < LDW) Bp[i] = (i < HID) ? b[i] : 0.0f;
}

// ---------------- layer 0: outer product [N,1]x[1,200], relu, write padded f16 ----------------

__global__ void k_l0(const float* __restrict__ agg, const float* __restrict__ W0,
                     const float* __restrict__ b0, _Float16* __restrict__ H, int n) {
  long long idx = (long long)blockIdx.x * blockDim.x + threadIdx.x;
  if (idx < (long long)n * LDH) {
    int node = (int)(idx / LDH);
    int j    = (int)(idx % LDH);
    float v = 0.0f;
    if (j < HID) {
      v = agg[node] * W0[j] + b0[j];
      v = v > 0.0f ? v : 0.0f;
    }
    H[idx] = (_Float16)v;                   // padding lanes write exact 0
  }
}

// ---------------- middle layers: [N,200] @ [200,200] + b, relu, in-place via WMMA ----------------
// 1024-thread block = 32 waves; weights staged in LDS (two 112-column panels, async copy),
// each wave owns 16 rows: A preloaded into 56 VGPRs, 91 v_wmma_f32_16x16x32_f16 per wave.

__global__ __launch_bounds__(1024) void k_mid(_Float16* __restrict__ H,
                                              const _Float16* __restrict__ Wp,
                                              const float* __restrict__ Bp, int n) {
  extern __shared__ _Float16 sW[];          // [LDH][SPLIT] halves

  int tid   = threadIdx.x;
  int lane  = tid & 31;
  int wavei = tid >> 5;                     // 0..31
  long long row0 = ((long long)blockIdx.x * 32 + wavei) * 16;
  bool active = (row0 + 16) <= (long long)n;  // wave-uniform

  int m  = lane & 15;                       // node within 16-row tile
  int kh = (lane >> 4) & 1;                 // 0: K {0-7,16-23}; 1: K {8-15,24-31}

  v16h a[KT];
  if (active) {
    const _Float16* Ap = H + (row0 + m) * LDH + kh * 8;
    #pragma unroll
    for (int kt = 0; kt < KT; ++kt) {
      v8h lo = *(const v8h*)(Ap + kt * 32);
      v8h hi = *(const v8h*)(Ap + kt * 32 + 16);
      a[kt] = __builtin_shufflevector(lo, hi, 0, 1, 2, 3, 4, 5, 6, 7,
                                              8, 9, 10, 11, 12, 13, 14, 15);
    }
  }

  #pragma unroll 1
  for (int s = 0; s < 2; ++s) {
    __syncthreads();                        // previous panel fully consumed
    // cooperative async copy of weight panel: rows 0..223, cols [s*112, s*112+cw)
    int cw  = s ? (LDW - SPLIT) : SPLIT;    // 112 then 96 halves
    int cpr = cw >> 3;                      // 16B chunks per row: 14 then 12
    int nch = LDH * cpr;
    for (int c = tid; c < nch; c += 1024) {
      int r = c / cpr, q = c - r * cpr;
      copy16_to_lds(Wp + (long long)r * LDW + s * SPLIT + q * 8,
                    sW + r * SPLIT + q * 8);
    }
    wait_async0();
    __syncthreads();

    if (active) {
      int nts = s ? (NT - 7) : 7;           // 7 n-tiles then 6
      #pragma unroll 1
      for (int ntl = 0; ntl < nts; ++ntl) {
        int nt = s * 7 + ntl;
        float bb = Bp[nt * 16 + m];
        v8f c = {bb, bb, bb, bb, bb, bb, bb, bb};
        #pragma unroll
        for (int kt = 0; kt < KT; ++kt) {
          // B tile from LDS: lane column n = m, elements over K in table order (kh*8 + {0-7,16-23})
          v16h bm;
          #pragma unroll
          for (int e = 0; e < 16; ++e)
            bm[e] = sW[(kt * 32 + kh * 8 + e + (e >= 8 ? 8 : 0)) * SPLIT + ntl * 16 + m];
          c = __builtin_amdgcn_wmma_f32_16x16x32_f16(false, a[kt], false, bm,
                                                     (short)0, c, false, false);
        }
        _Float16* op = H + (row0 + kh * 8) * LDH + nt * 16 + m;
        #pragma unroll
        for (int r = 0; r < 8; ++r) {
          float v = c[r];
          op[(long long)r * LDH] = (_Float16)(v > 0.0f ? v : 0.0f);
        }
      }
    }
  }
}

// ---------------- layer 7: dot([200]) + bias, sigmoid. One wave per node ----------------

__global__ void k_l7(const _Float16* __restrict__ H, const float* __restrict__ W7,
                     const float* __restrict__ b7, float* __restrict__ out, int n) {
  int lane = threadIdx.x & 31;
  int node = blockIdx.x * (blockDim.x >> 5) + (threadIdx.x >> 5);
  if (node >= n) return;
  const _Float16* h = H + (long long)node * LDH;
  float s = 0.0f;
  for (int k = lane; k < HID; k += 32) s += (float)h[k] * W7[k];
  #pragma unroll
  for (int off = 16; off > 0; off >>= 1) s += __shfl_xor(s, off, 32);
  if (lane == 0) out[node] = 1.0f / (1.0f + expf(-(s + b7[0])));
}

// ---------------- driver ----------------

extern "C" void kernel_launch(void* const* d_in, const int* in_sizes, int n_in,
                              void* d_out, int out_size, void* d_ws, size_t ws_size,
                              hipStream_t stream) {
  const float*     x  = (const float*)d_in[0];
  const long long* ei = (const long long*)d_in[1];
  const float* W[8];
  const float* B[8];
  for (int i = 0; i < 8; ++i) {
    W[i] = (const float*)d_in[2 + 2 * i];
    B[i] = (const float*)d_in[3 + 2 * i];
  }
  int n = in_sizes[0];          // 262144 nodes
  int E = in_sizes[1] / 2;      // 2097152 edges
  float* out = (float*)d_out;

  // workspace carve-up (d_ws >=256B aligned; all region sizes are 32B multiples)
  float*     deg = (float*)d_ws;
  float*     inv = deg + n;
  float*     agg = inv + n;
  float*     Bp  = agg + n;                           // 6 * LDW floats (4992 B)
  _Float16*  Wp  = (_Float16*)(Bp + 6 * LDW);         // 6 * LDH*LDW halves (559104 B)
  _Float16*  H   = Wp + 6 * LDH * LDW;                // n * LDH halves (~117 MB)

  const int T = 256;

  // GCN aggregate
  k_deg_init<<<(n + T - 1) / T, T, 0, stream>>>(deg, n);
  k_count   <<<(E + T - 1) / T, T, 0, stream>>>(ei, deg, E);
  k_norm    <<<(n + T - 1) / T, T, 0, stream>>>(x, deg, inv, agg, n);
  k_edge    <<<(E + T - 1) / T, T, 0, stream>>>(ei, x, inv, agg, E);

  // pad + convert weights for the 6 square layers
  for (int i = 0; i < 6; ++i) {
    k_prep_w<<<(LDH * LDW + T - 1) / T, T, 0, stream>>>(W[1 + i], Wp + (long long)i * LDH * LDW);
    k_prep_b<<<1, LDW, 0, stream>>>(B[1 + i], Bp + i * LDW);
  }

  // layer 0 (outer product) into padded f16 H
  long long tot0 = (long long)n * LDH;
  k_l0<<<(unsigned)((tot0 + T - 1) / T), T, 0, stream>>>(agg, W[0], B[0], H, n);

  // 6 WMMA layers, in place: 32 waves/block, weight panels staged in LDS
  int rows_per_block = 32 * 16;                       // 512
  int blocks = (n + rows_per_block - 1) / rows_per_block;
  size_t lds_bytes = (size_t)LDH * SPLIT * sizeof(_Float16);   // 50176
  for (int i = 0; i < 6; ++i)
    k_mid<<<blocks, 1024, lds_bytes, stream>>>(H, Wp + (long long)i * LDH * LDW,
                                               Bp + i * LDW, n);

  // final layer + sigmoid
  k_l7<<<(n + 7) / 8, 256, 0, stream>>>(H, W[7], B[7], out, n);
}
